// Net_LSV_18004502904848
// MI455X (gfx1250) — matrix operand order repaired and checked by reading, not compiled, not run on hardware
//
#include <hip/hip_runtime.h>
#include <hip/hip_bf16.h>
#include <math.h>

// ---------------------------------------------------------------------------
// Types
// ---------------------------------------------------------------------------
typedef _Float16 f16;
typedef __attribute__((ext_vector_type(16))) _Float16 v16h;
typedef __attribute__((ext_vector_type(8)))  _Float16 v8h;
typedef __attribute__((ext_vector_type(8)))  float    v8f;

// ---------------------------------------------------------------------------
// Problem constants
// ---------------------------------------------------------------------------
#define MC      65536
#define IND_T   64
#define PERIOD  16
#define N_MAT   4
#define N_K     21      // strikes

// ---------------------------------------------------------------------------
// Workspace layout (all offsets compile-time constants)
// Weights are stored padded f16, row-major [Kpad][Npad] per maturity index.
// ---------------------------------------------------------------------------
// f16 weight region (offsets in halves)
#define OFF_SVOL_L1 0            // 4 * 32*128
#define OFF_SVOL_L2 16384        // 4 * 128*128
#define OFF_SVOL_L3 81920        // 4 * 128*128
#define OFF_SVOL_L4 147456       // 4 * 128*16
#define OFF_VH_L1   155648       // 4 * 32*32
#define OFF_VH_L2   159744
#define OFF_VH_L3   163840
#define OFF_VD_L1   167936
#define OFF_VD_L2   172032
#define OFF_VD_L3   176128       // 4 * 32*16
#define OFF_VV_L1   178176
#define OFF_VV_L2   182272
#define OFF_VV_L3   186368       // 4 * 32*16
#define OFF_EH_L1   188416
#define OFF_EH_L2   192512
#define OFF_EH_L3   196608       // 4 * 32*16
#define W_TOTAL_HALVES 198656
#define WBYTES (W_TOTAL_HALVES * 2)    // 397312

// f32 bias region (offsets in floats)
#define BOFF_SVOL_B1 0
#define BOFF_SVOL_B2 512
#define BOFF_SVOL_B3 1024
#define BOFF_SVOL_B4 1536
#define BOFF_VH_B1   1600
#define BOFF_VH_B2   1728
#define BOFF_VH_B3   1856
#define BOFF_VD_B1   1984
#define BOFF_VD_B2   2112
#define BOFF_VD_B3   2240
#define BOFF_VV_B1   2304
#define BOFF_VV_B2   2432
#define BOFF_VV_B3   2560
#define BOFF_EH_B1   2624
#define BOFF_EH_B2   2752
#define BOFF_EH_B3   2880
#define B_TOTAL_FLOATS 2944
#define BBYTES (B_TOTAL_FLOATS * 4)    // 11776

// accumulator region (floats): [0..83] pv sum, [84..167] pv sumsq,
// [168] pe_u sum, [169] pe_u sq, [170] pe_h sum, [171] pe_h sq,
// [172] martingale sum, [173] put sum, [174] call sum
#define ACC_FLOATS 176
#define ACC_BYTE_OFF (WBYTES + BBYTES)

// ---------------------------------------------------------------------------
// Helpers
// ---------------------------------------------------------------------------
__device__ __forceinline__ float softplusf(float x) {
    return fmaxf(x, 0.f) + log1pf(expf(-fabsf(x)));
}

__device__ __forceinline__ float wave_sum(float v) {
    v += __shfl_xor(v, 16, 32);
    v += __shfl_xor(v,  8, 32);
    v += __shfl_xor(v,  4, 32);
    v += __shfl_xor(v,  2, 32);
    v += __shfl_xor(v,  1, 32);
    return v;
}

// Wait for this wave's outstanding LDS ops before cross-lane LDS reuse.
__device__ __forceinline__ void wait_ds() {
    asm volatile("s_wait_dscnt 0" ::: "memory");
}

// A-fragment (16x32 f16, M rows = paths) from LDS, row-major [16][KP].
// ISA layout: lanes 0-15: halves0-7 = K0..7, halves8-15 = K16..23 (M=lane);
//             lanes16-31: halves0-7 = K8..15, halves8-15 = K24..31 (M=lane-16)
__device__ __forceinline__ v16h load_a(const f16* act, int KP, int kbase) {
    int lane = threadIdx.x & 31;
    int M = lane & 15, hi = lane >> 4;
    const f16* p = act + M * KP + kbase + hi * 8;
    union { v16h v; v8h h[2]; } u;
    u.h[0] = *(const v8h*)p;
    u.h[1] = *(const v8h*)(p + 16);
    return u.v;
}

// B-fragment (32x16 f16, K x N): N = lane&15; halves0-7 = K 8*hi+0..7, halves8-15 = K 16+8*hi+0..7.
// Weights stored f16 row-major [KP][NPW], read straight from global (L2-hot).
__device__ __forceinline__ v16h load_b(const f16* W, int NPW, int kbase, int nbase) {
    int lane = threadIdx.x & 31;
    int hi = lane >> 4;
    const f16* p = W + (kbase + hi * 8) * NPW + nbase + (lane & 15);
    v16h v;
#pragma unroll
    for (int s = 0; s < 8; ++s) {
        v[s] = p[s * NPW];
        v[s + 8] = p[(16 + s) * NPW];
    }
    return v;
}

// Generic WMMA layer: actIn [16][KP] (LDS, f16) x W [KP][NPW] -> actOut [16][outStride]
// mode: 0 = linear, 1 = relu, 2 = softplus.   NP multiple of 16, KP multiple of 32.
__device__ __forceinline__ void layer(const f16* actIn, int KP,
                                      const f16* W, int NPW,
                                      const float* bias, int NP,
                                      f16* actOut, int outStride, int mode) {
    int lane = threadIdx.x & 31;
    int colL = lane & 15, hi = lane >> 4;
    for (int nt = 0; nt < NP / 16; ++nt) {
        v8f c = {0.f, 0.f, 0.f, 0.f, 0.f, 0.f, 0.f, 0.f};
        for (int ks = 0; ks < KP / 32; ++ks) {
            v16h a = load_a(actIn, KP, ks * 32);
            v16h b = load_b(W, NPW, ks * 32, nt * 16);
            c = __builtin_amdgcn_wmma_f32_16x16x32_f16(false, a, false, b,
                                                       (short)0, c, false, false);
        }
        int col = nt * 16 + colL;
        float bv = bias[col];
#pragma unroll
        for (int vg = 0; vg < 8; ++vg) {
            float x = c[vg] + bv;
            if (mode == 1) x = fmaxf(x, 0.f);
            else if (mode == 2) x = softplusf(x);
            actOut[(vg + 8 * hi) * outStride + col] = (f16)x;
        }
    }
    wait_ds();
}

// Layer with scalar (N=1) output: W is [KP][16] with only column 0 valid.
// Result per path delivered through out16 LDS; returned in all lanes (path = lane&15).
__device__ __forceinline__ float layer_scalar(const f16* actIn, int KP,
                                              const f16* W, const float* bias,
                                              float* out16, int mode, int path) {
    int lane = threadIdx.x & 31;
    int colL = lane & 15, hi = lane >> 4;
    v8f c = {0.f, 0.f, 0.f, 0.f, 0.f, 0.f, 0.f, 0.f};
    for (int ks = 0; ks < KP / 32; ++ks) {
        v16h a = load_a(actIn, KP, ks * 32);
        v16h b = load_b(W, 16, ks * 32, 0);
        c = __builtin_amdgcn_wmma_f32_16x16x32_f16(false, a, false, b,
                                                   (short)0, c, false, false);
    }
    if (colL == 0) {
        float bv = bias[0];
#pragma unroll
        for (int vg = 0; vg < 8; ++vg) {
            float x = c[vg] + bv;
            if (mode == 2) x = softplusf(x);
            out16[vg + 8 * hi] = x;
        }
    }
    wait_ds();
    return out16[path];
}

// vh output layer fused with control-variate accumulation:
// cvM[path][k] += softplus(out[path][k]) * dS[path]   (k < 21)
__device__ __forceinline__ void layer_vh_out(const f16* actIn, const f16* W,
                                             const float* bias,
                                             float* cvM, const float* dSb) {
    int lane = threadIdx.x & 31;
    int colL = lane & 15, hi = lane >> 4;
    for (int nt = 0; nt < 2; ++nt) {
        v8f c = {0.f, 0.f, 0.f, 0.f, 0.f, 0.f, 0.f, 0.f};
        v16h a = load_a(actIn, 32, 0);
        v16h b = load_b(W, 32, 0, nt * 16);
        c = __builtin_amdgcn_wmma_f32_16x16x32_f16(false, a, false, b,
                                                   (short)0, c, false, false);
        int col = nt * 16 + colL;
        if (col < N_K) {
            float bv = bias[col];
#pragma unroll
            for (int vg = 0; vg < 8; ++vg) {
                int row = vg + 8 * hi;
                float v = softplusf(c[vg] + bv);
                cvM[row * N_K + col] += v * dSb[row];
            }
        }
    }
    wait_ds();
}

// Write one MLP input row (<=3 features) into an LDS [16][32] buffer, zero-padded.
__device__ __forceinline__ void write_in3(f16* buf, int path, bool active,
                                          float a, float b, float c) {
    if (active) {
        v8h z;
#pragma unroll
        for (int q = 0; q < 8; ++q) z[q] = (f16)0.f;
        v8h first = z;
        first[0] = (f16)a; first[1] = (f16)b; first[2] = (f16)c;
        v8h* r = (v8h*)(buf + path * 32);
        r[0] = first; r[1] = z; r[2] = z; r[3] = z;
    }
    wait_ds();
}

// ---------------------------------------------------------------------------
// Prep kernels: pad/convert weights (f32 -> padded f16 tiles) and biases.
// ---------------------------------------------------------------------------
__global__ void pad_weights(const float* __restrict__ W, f16* __restrict__ dst,
                            int nIdx, int dIn, int dOut, int KP, int NP) {
    int tot = nIdx * KP * NP;
    for (int e = blockIdx.x * blockDim.x + threadIdx.x; e < tot;
         e += gridDim.x * blockDim.x) {
        int idx = e / (KP * NP);
        int rem = e % (KP * NP);
        int r = rem / NP, c = rem % NP;
        float v = (r < dIn && c < dOut) ? W[(idx * dIn + r) * dOut + c] : 0.f;
        dst[e] = (f16)v;
    }
}

__global__ void pad_bias(const float* __restrict__ b, float* __restrict__ dst,
                         int nIdx, int dOut, int NP) {
    int tot = nIdx * NP;
    for (int e = blockIdx.x * blockDim.x + threadIdx.x; e < tot;
         e += gridDim.x * blockDim.x) {
        int idx = e / NP, c = e % NP;
        dst[e] = (c < dOut) ? b[idx * dOut + c] : 0.f;
    }
}

__global__ void zero_acc(float* acc) {
    int t = blockIdx.x * blockDim.x + threadIdx.x;
    if (t < ACC_FLOATS) acc[t] = 0.f;
}

// ---------------------------------------------------------------------------
// Main simulation kernel: 4 waves/block, 16 paths/wave.
// ---------------------------------------------------------------------------
#define WAVES 4

struct alignas(16) WaveShared {
    f16   actA[16 * 128];          // 4 KB
    f16   actB[16 * 128];          // 4 KB
    float cv[N_MAT][16][N_K];      // 5.25 KB
    float out16[16];
    float dS[16];
};

__global__ __launch_bounds__(WAVES * 32) void lsv_sim(
    const float* __restrict__ S0p, const float* __restrict__ ratep,
    const float* __restrict__ z,   const float* __restrict__ z2,
    const float* __restrict__ tg,  const float* __restrict__ strikes,
    const float* __restrict__ v0p, const float* __restrict__ rhop,
    const f16* __restrict__ wts,   const float* __restrict__ biases,
    float* acc) {
    __shared__ WaveShared sm[WAVES];
    int wave = threadIdx.x >> 5, lane = threadIdx.x & 31;
    WaveShared& WS = sm[wave];
    int path = lane & 15;
    bool act = lane < 16;
    int gpath = (blockIdx.x * WAVES + wave) * 16 + path;

    // zero activation buffers (pad columns must stay 0) and cv accumulators
    {
        unsigned* pA = (unsigned*)WS.actA;
        unsigned* pB = (unsigned*)WS.actB;
        for (int e = lane; e < 1024; e += 32) { pA[e] = 0u; pB[e] = 0u; }
        float* cv = &WS.cv[0][0][0];
        for (int e = lane; e < N_MAT * 16 * N_K; e += 32) cv[e] = 0.f;
    }
    wait_ds();

    float S0 = S0p[0], rate = ratep[0];
    float rho = tanhf(rhop[0]);
    float rho2c = sqrtf(fmaxf(1.f - rho * rho, 0.f));
    float S = S0;
    float V = (1.f / (1.f + expf(-v0p[0]))) * 0.5f;
    float runmax = S;
    float cvex = 0.f;

    f16* smallA = WS.actA;   // reused as [16][32] for the 20-wide nets
    f16* smallB = WS.actB;

#pragma unroll 1
    for (int i = 1; i <= IND_T; ++i) {
        int idx = (i - 1) >> 4;
        float t0 = tg[i - 1], t1 = tg[i];
        float h = t1 - t0, sqh = sqrtf(h);
        float zz  = act ? z [gpath * IND_T + (i - 1)] : 0.f;
        float zz2 = act ? z2[gpath * IND_T + (i - 1)] : 0.f;
        float dW = sqh * zz;
        float dB = rho * dW + rho2c * sqh * zz2;

        // ---- svol: [t,S,V] -> 100 -> 100 -> 100 -> softplus(1) ----
        write_in3(smallA, path, act, t0, S, V);
        layer(smallA, 32, wts + OFF_SVOL_L1 + idx * 32 * 128, 128,
              biases + BOFF_SVOL_B1 + idx * 128, 112, WS.actB, 128, 1);
        layer(WS.actB, 128, wts + OFF_SVOL_L2 + idx * 128 * 128, 128,
              biases + BOFF_SVOL_B2 + idx * 128, 112, WS.actA, 128, 1);
        layer(WS.actA, 128, wts + OFF_SVOL_L3 + idx * 128 * 128, 128,
              biases + BOFF_SVOL_B3 + idx * 128, 112, WS.actB, 128, 1);
        float pd = layer_scalar(WS.actB, 128, wts + OFF_SVOL_L4 + idx * 128 * 16,
                                biases + BOFF_SVOL_B4 + idx * 16, WS.out16, 2, path);
        float price_diff = S * pd;

        // ---- eh: [t,S] -> 20 -> 20 -> 1 (linear) ----
        write_in3(smallA, path, act, t0, S, 0.f);
        layer(smallA, 32, wts + OFF_EH_L1 + idx * 1024, 32,
              biases + BOFF_EH_B1 + idx * 32, 32, smallB, 32, 1);
        layer(smallB, 32, wts + OFF_EH_L2 + idx * 1024, 32,
              biases + BOFF_EH_B2 + idx * 32, 32, smallA, 32, 1);
        float ehv = layer_scalar(smallA, 32, wts + OFF_EH_L3 + idx * 512,
                                 biases + BOFF_EH_B3 + idx * 16, WS.out16, 0, path);

        // ---- vd: [V] -> 20 -> 20 -> 1 (linear) ----
        write_in3(smallA, path, act, V, 0.f, 0.f);
        layer(smallA, 32, wts + OFF_VD_L1 + idx * 1024, 32,
              biases + BOFF_VD_B1 + idx * 32, 32, smallB, 32, 1);
        layer(smallB, 32, wts + OFF_VD_L2 + idx * 1024, 32,
              biases + BOFF_VD_B2 + idx * 32, 32, smallA, 32, 1);
        float vdv = layer_scalar(smallA, 32, wts + OFF_VD_L3 + idx * 512,
                                 biases + BOFF_VD_B3 + idx * 16, WS.out16, 0, path);

        // ---- vv: [V] -> 20 -> 20 -> softplus(1) ----
        write_in3(smallA, path, act, V, 0.f, 0.f);
        layer(smallA, 32, wts + OFF_VV_L1 + idx * 1024, 32,
              biases + BOFF_VV_B1 + idx * 32, 32, smallB, 32, 1);
        layer(smallB, 32, wts + OFF_VV_L2 + idx * 1024, 32,
              biases + BOFF_VV_B2 + idx * 32, 32, smallA, 32, 1);
        float vvv = layer_scalar(smallA, 32, wts + OFF_VV_L3 + idx * 512,
                                 biases + BOFF_VV_B3 + idx * 16, WS.out16, 2, path);

        // ---- state update (tamed Euler) ----
        float V_new = V + vdv * h + vvv * dB;
        float drift = rate * S;
        float diff  = price_diff * dW;
        float drift_c = 1.f + fabsf(drift) * sqh;
        float diff_c  = 1.f + fabsf(price_diff) * sqh;
        float S_new = S + drift * h / drift_c + diff / diff_c;
        float dS = expf(-rate * t1) * S_new - expf(-rate * t0) * S;
        if (act) WS.dS[path] = dS;
        wait_ds();
        cvex += ehv * dS;

        // ---- vh for m = idx..3: [t,S_old] -> 20 -> 20 -> softplus(21), fused cv += ----
#pragma unroll 1
        for (int m = idx; m < N_MAT; ++m) {
            write_in3(smallA, path, act, t0, S, 0.f);
            layer(smallA, 32, wts + OFF_VH_L1 + m * 1024, 32,
                  biases + BOFF_VH_B1 + m * 32, 32, smallB, 32, 1);
            layer(smallB, 32, wts + OFF_VH_L2 + m * 1024, 32,
                  biases + BOFF_VH_B2 + m * 32, 32, smallA, 32, 1);
            layer_vh_out(smallA, wts + OFF_VH_L3 + m * 1024,
                         biases + BOFF_VH_B3 + m * 32, &WS.cv[m][0][0], WS.dS);
        }

        runmax = fmaxf(runmax, S_new);
        S = S_new;
        V = fmaxf(V_new, 0.f);

        // ---- maturity statistics ----
        if ((i & (PERIOD - 1)) == 0) {
            float dis = expf(-rate * t1);
            for (int k = 0; k < N_K; ++k) {
                float v = 0.f;
                if (act) {
                    float payoff = fmaxf(S - strikes[k], 0.f);
                    v = dis * payoff - WS.cv[idx][path][k];
                }
                float s1 = wave_sum(v);
                float s2 = wave_sum(v * v);
                if (lane == 0) {
                    atomicAdd(&acc[idx * N_K + k], s1);
                    atomicAdd(&acc[84 + idx * N_K + k], s2);
                }
            }
            if (i == IND_T) {
                float expay = runmax - S;
                float peu  = act ? dis * expay : 0.f;
                float peh  = act ? (dis * expay - cvex) : 0.f;
                float mart = act ? dis * S : 0.f;
                float put  = act ? dis * fmaxf(S0 - S, 0.f) : 0.f;
                float call = act ? dis * fmaxf(S - S0, 0.f) : 0.f;
                float r0 = wave_sum(peu),  r1 = wave_sum(peu * peu);
                float r2 = wave_sum(peh),  r3 = wave_sum(peh * peh);
                float r4 = wave_sum(mart), r5 = wave_sum(put), r6 = wave_sum(call);
                if (lane == 0) {
                    atomicAdd(&acc[168], r0); atomicAdd(&acc[169], r1);
                    atomicAdd(&acc[170], r2); atomicAdd(&acc[171], r3);
                    atomicAdd(&acc[172], r4); atomicAdd(&acc[173], r5);
                    atomicAdd(&acc[174], r6);
                }
            }
        }
    }
}

// ---------------------------------------------------------------------------
// Finalize: sums -> means / unbiased variances, reference output order.
// ---------------------------------------------------------------------------
__global__ void finalize(const float* __restrict__ acc,
                         const float* __restrict__ S0p,
                         const float* __restrict__ ratep,
                         const float* __restrict__ tg, float* out) {
    int t = threadIdx.x;
    const float MCf = (float)MC;
    if (t < 84) {
        float s = acc[t], sq = acc[84 + t];
        out[t]      = s / MCf;
        out[84 + t] = (sq - s * s / MCf) / (MCf - 1.f);
    }
    if (t == 0) {
        float S0 = S0p[0], rate = ratep[0];
        float dis = expf(-rate * tg[IND_T]);
        float peu_s = acc[168], peu_q = acc[169];
        float peh_s = acc[170], peh_q = acc[171];
        float mart = acc[172], put = acc[173], call = acc[174];
        out[168] = peu_s / MCf;                                   // pe_u
        out[169] = peh_s / MCf;                                   // pe_h
        out[170] = (peu_q - peu_s * peu_s / MCf) / (MCf - 1.f);   // pe_var_u
        out[171] = (peh_q - peh_s * peh_s / MCf) / (MCf - 1.f);   // pe_var_h
        out[172] = mart / (MCf * S0);                             // martingale_test
        out[173] = put / MCf;                                     // put_atm
        out[174] = call / MCf;                                    // call_atm
        out[175] = call / MCf - put / MCf - S0 + dis * S0;        // pcp_err
    }
}

// ---------------------------------------------------------------------------
// Launch
// d_in order (setup_inputs insertion order, params flattened svol,vh,vd,vv,eh
// with Ws then bs per net):
//  0 S0, 1 rate, 2 z, 3 z2, 4 timegrid, 5 strikes,
//  6..9  svol W0..W3, 10..13 svol b0..b3,
// 14..16 vh W0..W2, 17..19 vh b0..b2,
// 20..22 vd W, 23..25 vd b, 26..28 vv W, 29..31 vv b, 32..34 eh W, 35..37 eh b,
// 38 v0, 39 rho, 40.. ints (unused)
// ---------------------------------------------------------------------------
extern "C" void kernel_launch(void* const* d_in, const int* in_sizes, int n_in,
                              void* d_out, int out_size, void* d_ws, size_t ws_size,
                              hipStream_t stream) {
    if (ws_size < (size_t)(ACC_BYTE_OFF + ACC_FLOATS * 4)) return;

    char* wsb = (char*)d_ws;
    f16*   wts    = (f16*)wsb;
    float* biases = (float*)(wsb + WBYTES);
    float* acc    = (float*)(wsb + ACC_BYTE_OFF);

#define PADW(ii, off, dIn, dOut, KP, NP)                                        \
    pad_weights<<<(4 * (KP) * (NP) + 255) / 256, 256, 0, stream>>>(             \
        (const float*)d_in[ii], wts + (off), 4, dIn, dOut, KP, NP)
#define PADB(ii, off, dOut, NP)                                                 \
    pad_bias<<<(4 * (NP) + 255) / 256, 256, 0, stream>>>(                       \
        (const float*)d_in[ii], biases + (off), 4, dOut, NP)

    // svol
    PADW(6,  OFF_SVOL_L1,   3, 100,  32, 128);
    PADW(7,  OFF_SVOL_L2, 100, 100, 128, 128);
    PADW(8,  OFF_SVOL_L3, 100, 100, 128, 128);
    PADW(9,  OFF_SVOL_L4, 100,   1, 128,  16);
    PADB(10, BOFF_SVOL_B1, 100, 128);
    PADB(11, BOFF_SVOL_B2, 100, 128);
    PADB(12, BOFF_SVOL_B3, 100, 128);
    PADB(13, BOFF_SVOL_B4,   1,  16);
    // vh
    PADW(14, OFF_VH_L1,  2, 20, 32, 32);
    PADW(15, OFF_VH_L2, 20, 20, 32, 32);
    PADW(16, OFF_VH_L3, 20, 21, 32, 32);
    PADB(17, BOFF_VH_B1, 20, 32);
    PADB(18, BOFF_VH_B2, 20, 32);
    PADB(19, BOFF_VH_B3, 21, 32);
    // vd
    PADW(20, OFF_VD_L1,  1, 20, 32, 32);
    PADW(21, OFF_VD_L2, 20, 20, 32, 32);
    PADW(22, OFF_VD_L3, 20,  1, 32, 16);
    PADB(23, BOFF_VD_B1, 20, 32);
    PADB(24, BOFF_VD_B2, 20, 32);
    PADB(25, BOFF_VD_B3,  1, 16);
    // vv
    PADW(26, OFF_VV_L1,  1, 20, 32, 32);
    PADW(27, OFF_VV_L2, 20, 20, 32, 32);
    PADW(28, OFF_VV_L3, 20,  1, 32, 16);
    PADB(29, BOFF_VV_B1, 20, 32);
    PADB(30, BOFF_VV_B2, 20, 32);
    PADB(31, BOFF_VV_B3,  1, 16);
    // eh
    PADW(32, OFF_EH_L1,  2, 20, 32, 32);
    PADW(33, OFF_EH_L2, 20, 20, 32, 32);
    PADW(34, OFF_EH_L3, 20,  1, 32, 16);
    PADB(35, BOFF_EH_B1, 20, 32);
    PADB(36, BOFF_EH_B2, 20, 32);
    PADB(37, BOFF_EH_B3,  1, 16);
#undef PADW
#undef PADB

    zero_acc<<<1, 256, 0, stream>>>(acc);

    // 16 paths/wave * 4 waves/block = 64 paths/block -> 1024 blocks
    lsv_sim<<<MC / (WAVES * 16), WAVES * 32, 0, stream>>>(
        (const float*)d_in[0], (const float*)d_in[1],
        (const float*)d_in[2], (const float*)d_in[3],
        (const float*)d_in[4], (const float*)d_in[5],
        (const float*)d_in[38], (const float*)d_in[39],
        wts, biases, acc);

    finalize<<<1, 256, 0, stream>>>(acc, (const float*)d_in[0],
                                    (const float*)d_in[1],
                                    (const float*)d_in[4], (float*)d_out);
}
